// ImprovedDynamicCellVQVAE_16423954940666
// MI455X (gfx1250) — compile-verified, matrix operand order checked by reading, not run on hardware
//
#include <hip/hip_runtime.h>
#include <math.h>

// ---------------------------------------------------------------------------
// CDNA5 (gfx1250) wave32 WMMA bf16 GEMM/conv core, double-buffered LDS
// ---------------------------------------------------------------------------
typedef __attribute__((ext_vector_type(16))) __bf16 v16bf;
typedef __attribute__((ext_vector_type(8)))  __bf16 v8bf;
typedef __attribute__((ext_vector_type(8)))  float  v8f;

#define BM 32
#define BN 128          // block N tile; each of 8 waves owns 16(M) x 32(N)
#define BK 32
#define NT 256
#define APITCH 40       // LDS row pitch in bf16 elems (80B: 16B-aligned, conflict-spread)
#define ASLOTS ((BM * 16) / NT)   // 2 staging pair-slots per thread for A
#define BSLOTS ((BN * 16) / NT)   // 8 staging pair-slots per thread for B

static __device__ __forceinline__ int imin_(int a, int b) { return a < b ? a : b; }
static __device__ __forceinline__ int imax_(int a, int b) { return a > b ? a : b; }

// RNE fp32 -> bf16 pair packed into u32 (lo in [15:0], hi in [31:16])
static __device__ __forceinline__ unsigned pack2bf(float a, float b) {
  union { float f; unsigned u; } x, y;
  x.f = a; y.f = b;
  unsigned ua = (x.u + 0x7FFFu + ((x.u >> 16) & 1u)) >> 16;
  unsigned ub = (y.u + 0x7FFFu + ((y.u >> 16) & 1u)) >> 16;
  return (ub << 16) | (ua & 0xFFFFu);
}

// K index of fragment element e for the ISA 16-bit 16x32 layout
static __device__ __forceinline__ int frag_k(int e, int hi) {
  int p = e >> 1;
  int base = (p < 4) ? (2 * p) : (16 + 2 * (p - 4));
  return base + hi * 8 + (e & 1);
}

// 32B fragment load from LDS as 2 x ds_load_b128
static __device__ __forceinline__ v16bf ld_frag(const __bf16* p) {
  v8bf lo = *(const v8bf*)p;
  v8bf hi = *(const v8bf*)(p + 8);
  return __builtin_shufflevector(lo, hi, 0, 1, 2, 3, 4, 5, 6, 7,
                                 8, 9, 10, 11, 12, 13, 14, 15);
}

// ---------------------------------------------------------------------------
// Generic strided batched GEMM:  Out[b,m,n] = alpha * sum_k A[b,m,k]*B[b,k,n]
//                                            * scale[m] + shift[m] (+resid) (relu)
// fp32 sources -> bf16 staged to double-buffered LDS in fragment order.
// ---------------------------------------------------------------------------
__global__ __launch_bounds__(256)
void gemm_bf16_wmma(const float* __restrict__ A, long a_bs, long a_sm, long a_sk,
                    const float* __restrict__ Bsrc, long b_bs, long b_sk, long b_sn,
                    float* __restrict__ Out, long o_bs, long o_sm, long o_sn,
                    const float* __restrict__ scale, const float* __restrict__ shift,
                    const float* __restrict__ residual, const float* __restrict__ alpha_ptr,
                    int M, int N, int K, int relu) {
  __shared__ __attribute__((aligned(32))) __bf16 lA[2][BM * APITCH];
  __shared__ __attribute__((aligned(32))) __bf16 lB[2][BN * APITCH];
  const int batch = blockIdx.z;
  const int tm = blockIdx.y * BM;
  const int tn = blockIdx.x * BN;
  const int tid = threadIdx.x;
  const int lane = tid & 31;
  const int wave = tid >> 5;
  const int wm = (wave >> 2) * 16;
  const int wn = (wave & 3) * 32;
  const int hi = lane >> 4;
  const int l15 = lane & 15;
  const float* Ab = A + (long)batch * a_bs;
  const float* Bb = Bsrc + (long)batch * b_bs;

  // ---- loop-invariant staging descriptors (per thread) ----
  int  aK[ASLOTS], aLidx[ASLOTS];  long aRow[ASLOTS];  bool aOk[ASLOTS];
#pragma unroll
  for (int it = 0; it < ASLOTS; ++it) {
    int i = tid + it * NT;
    int e2 = i & 7, h = (i >> 3) & 1, m = i >> 4;
    aK[it] = frag_k(e2 * 2, h);
    aOk[it] = (tm + m) < M;
    aRow[it] = (long)imin_(tm + m, M - 1) * a_sm;
    aLidx[it] = m * (APITCH / 2) + h * 8 + e2;
  }
  int  bK[BSLOTS], bLidx[BSLOTS];  long bCol[BSLOTS];  bool bOk[BSLOTS];
#pragma unroll
  for (int it = 0; it < BSLOTS; ++it) {
    int i = tid + it * NT;
    int e2 = i & 7, h = (i >> 3) & 1, n = i >> 4;
    bK[it] = frag_k(e2 * 2, h);
    bOk[it] = (tn + n) < N;
    bCol[it] = (long)imin_(tn + n, N - 1) * b_sn;
    bLidx[it] = n * (APITCH / 2) + h * 8 + e2;
  }

  auto stage = [&](int p, int k0) {
    unsigned* lAu = (unsigned*)lA[p];
    unsigned* lBu = (unsigned*)lB[p];
#pragma unroll
    for (int it = 0; it < ASLOTS; ++it) {
      int ka = k0 + aK[it];
      float va = Ab[aRow[it] + (long)imin_(ka, K - 1) * a_sk];
      float vb = Ab[aRow[it] + (long)imin_(ka + 1, K - 1) * a_sk];
      va = (aOk[it] && ka < K) ? va : 0.f;
      vb = (aOk[it] && ka + 1 < K) ? vb : 0.f;
      lAu[aLidx[it]] = pack2bf(va, vb);
    }
#pragma unroll
    for (int it = 0; it < BSLOTS; ++it) {
      int ka = k0 + bK[it];
      float va = Bb[(long)imin_(ka, K - 1) * b_sk + bCol[it]];
      float vb = Bb[(long)imin_(ka + 1, K - 1) * b_sk + bCol[it]];
      va = (bOk[it] && ka < K) ? va : 0.f;
      vb = (bOk[it] && ka + 1 < K) ? vb : 0.f;
      lBu[bLidx[it]] = pack2bf(va, vb);
    }
  };

  v8f acc0 = {}, acc1 = {};
  stage(0, 0);
  __syncthreads();
  int p = 0;
  for (int k0 = 0; k0 < K; k0 += BK) {
    v16bf af = ld_frag(&lA[p][(wm + l15) * APITCH + hi * 16]);
    v16bf b0 = ld_frag(&lB[p][(wn + l15) * APITCH + hi * 16]);
    v16bf b1 = ld_frag(&lB[p][(wn + 16 + l15) * APITCH + hi * 16]);
    acc0 = __builtin_amdgcn_wmma_f32_16x16x32_bf16(false, af, false, b0,
                                                   (short)0, acc0, false, false);
    acc1 = __builtin_amdgcn_wmma_f32_16x16x32_bf16(false, af, false, b1,
                                                   (short)0, acc1, false, false);
    if (k0 + BK < K) stage(p ^ 1, k0 + BK);   // overlap gather with matrix pipe
    __syncthreads();
    p ^= 1;
  }

  const float alpha = alpha_ptr ? *alpha_ptr : 1.0f;
#pragma unroll
  for (int half = 0; half < 2; ++half) {
    const v8f& acc = half ? acc1 : acc0;
    int n = tn + wn + half * 16 + l15;
    if (n < N) {
#pragma unroll
      for (int r = 0; r < 8; ++r) {
        int m = tm + wm + r + hi * 8;
        if (m < M) {
          float v = acc[r] * alpha;
          if (scale) v *= scale[m];
          if (shift) v += shift[m];
          long oi = (long)batch * o_bs + (long)m * o_sm + (long)n * o_sn;
          if (residual) v += residual[oi];
          if (relu) v = v > 0.f ? v : 0.f;
          Out[oi] = v;
        }
      }
    }
  }
}

// ---------------------------------------------------------------------------
// Implicit-GEMM convolution (also transposed conv via lhs_dilation + flipped
// weights).  M = OC, K = IC*KH*KW, N = NB*OH*OW.  Weights fp32 [OC, K] dense.
// Clamped unconditional gather + select; double-buffered LDS; 1x1 fast path.
// ---------------------------------------------------------------------------
__global__ __launch_bounds__(256)
void conv_bf16_wmma(const float* __restrict__ X, const float* __restrict__ Wt,
                    const float* __restrict__ scale, const float* __restrict__ shift,
                    const float* __restrict__ residual, float* __restrict__ Out,
                    int NB, int IC, int IH, int IW, int OC, int OH, int OW,
                    int KH, int KW, int stride, int pad, int ldil, int relu) {
  __shared__ __attribute__((aligned(32))) __bf16 lA[2][BM * APITCH];
  __shared__ __attribute__((aligned(32))) __bf16 lB[2][BN * APITCH];
  const int K = IC * KH * KW;
  const int N = NB * OH * OW;
  const int KHKW = KH * KW;
  const int OHOW = OH * OW;
  const int IHIW = IH * IW;
  const int tm = blockIdx.y * BM;
  const int tn = blockIdx.x * BN;
  const int tid = threadIdx.x;
  const int lane = tid & 31;
  const int wave = tid >> 5;
  const int wm = (wave >> 2) * 16;
  const int wn = (wave & 3) * 32;
  const int hi = lane >> 4;
  const int l15 = lane & 15;

  // ---- loop-invariant staging descriptors ----
  int  aK[ASLOTS], aLidx[ASLOTS];  long aRow[ASLOTS];  bool aOk[ASLOTS];
#pragma unroll
  for (int it = 0; it < ASLOTS; ++it) {
    int i = tid + it * NT;
    int e2 = i & 7, h = (i >> 3) & 1, m = i >> 4;
    aK[it] = frag_k(e2 * 2, h);
    aOk[it] = (tm + m) < OC;
    aRow[it] = (long)imin_(tm + m, OC - 1) * K;
    aLidx[it] = m * (APITCH / 2) + h * 8 + e2;
  }
  int  bK[BSLOTS], bLidx[BSLOTS], bOh[BSLOTS], bOw[BSLOTS];
  long bXoff[BSLOTS];  bool bOk[BSLOTS];
#pragma unroll
  for (int it = 0; it < BSLOTS; ++it) {
    int i = tid + it * NT;
    int e2 = i & 7, h = (i >> 3) & 1, n = i >> 4;
    int ng = tn + n;
    int ngc = imin_(ng, N - 1);
    int b = ngc / OHOW;
    int r2 = ngc - b * OHOW;
    bOh[it] = r2 / OW;
    bOw[it] = r2 - bOh[it] * OW;
    bXoff[it] = (long)b * IC * IHIW;
    bK[it] = frag_k(e2 * 2, h);
    bOk[it] = ng < N;
    bLidx[it] = n * (APITCH / 2) + h * 8 + e2;
  }

  auto stage = [&](int p, int k0) {
    unsigned* lAu = (unsigned*)lA[p];
    unsigned* lBu = (unsigned*)lB[p];
#pragma unroll
    for (int it = 0; it < ASLOTS; ++it) {
      int ka = k0 + aK[it];
      float va = Wt[aRow[it] + imin_(ka, K - 1)];
      float vb = Wt[aRow[it] + imin_(ka + 1, K - 1)];
      va = (aOk[it] && ka < K) ? va : 0.f;
      vb = (aOk[it] && ka + 1 < K) ? vb : 0.f;
      lAu[aLidx[it]] = pack2bf(va, vb);
    }
#pragma unroll
    for (int it = 0; it < BSLOTS; ++it) {
      float vv[2];
#pragma unroll
      for (int t = 0; t < 2; ++t) {
        int kg = k0 + bK[it] + t;
        int kgc = imin_(kg, K - 1);
        int ci, kh, kw;
        if (KHKW == 1) {           // uniform fast path: 1x1 convs
          ci = kgc; kh = 0; kw = 0;
        } else {
          ci = kgc / KHKW;
          int rem = kgc - ci * KHKW;
          kh = rem / KW;
          kw = rem - kh * KW;
        }
        int ihp = bOh[it] * stride - pad + kh;
        int iwp = bOw[it] * stride - pad + kw;
        bool ok = bOk[it] && (kg < K);
        int ih, iw;
        if (ldil == 1) {
          ih = ihp; iw = iwp;
        } else {
          ok = ok && (((ihp | iwp) & 1) == 0);
          ih = ihp >> 1; iw = iwp >> 1;
        }
        ok = ok && ((unsigned)ih < (unsigned)IH) && ((unsigned)iw < (unsigned)IW);
        int ihc = imin_(imax_(ih, 0), IH - 1);
        int iwc = imin_(imax_(iw, 0), IW - 1);
        float raw = X[bXoff[it] + (long)ci * IHIW + ihc * IW + iwc];
        vv[t] = ok ? raw : 0.f;
      }
      lBu[bLidx[it]] = pack2bf(vv[0], vv[1]);
    }
  };

  v8f acc0 = {}, acc1 = {};
  stage(0, 0);
  __syncthreads();
  int p = 0;
  for (int k0 = 0; k0 < K; k0 += BK) {
    // prefetch next-next weight tile (global_prefetch_b8 on gfx1250)
    if (tid < BM && k0 + 2 * BK < K) {
      int m = imin_(tm + tid, OC - 1);
      __builtin_prefetch(&Wt[(long)m * K + k0 + 2 * BK], 0, 1);
    }
    v16bf af = ld_frag(&lA[p][(wm + l15) * APITCH + hi * 16]);
    v16bf b0 = ld_frag(&lB[p][(wn + l15) * APITCH + hi * 16]);
    v16bf b1 = ld_frag(&lB[p][(wn + 16 + l15) * APITCH + hi * 16]);
    acc0 = __builtin_amdgcn_wmma_f32_16x16x32_bf16(false, af, false, b0,
                                                   (short)0, acc0, false, false);
    acc1 = __builtin_amdgcn_wmma_f32_16x16x32_bf16(false, af, false, b1,
                                                   (short)0, acc1, false, false);
    if (k0 + BK < K) stage(p ^ 1, k0 + BK);
    __syncthreads();
    p ^= 1;
  }

#pragma unroll
  for (int half = 0; half < 2; ++half) {
    const v8f& acc = half ? acc1 : acc0;
    int n = tn + wn + half * 16 + l15;
    if (n < N) {
      int b = n / OHOW;
      int rr = n - b * OHOW;
#pragma unroll
      for (int r = 0; r < 8; ++r) {
        int m = tm + wm + r + hi * 8;
        if (m < OC) {
          float v = acc[r];
          if (scale) v *= scale[m];
          if (shift) v += shift[m];
          long oi = ((long)b * OC + m) * OHOW + rr;
          if (residual) v += residual[oi];
          if (relu) v = v > 0.f ? v : 0.f;
          Out[oi] = v;
        }
      }
    }
  }
}

// ---------------------------------------------------------------------------
// Small helper kernels
// ---------------------------------------------------------------------------
__global__ void zero_kernel(float* p, int n) {
  int i = blockIdx.x * blockDim.x + threadIdx.x;
  if (i < n) p[i] = 0.f;
}

__global__ void fold_bn_kernel(const float* __restrict__ g, const float* __restrict__ bt,
                               const float* __restrict__ mn, const float* __restrict__ vr,
                               const float* __restrict__ bias, float* scale, float* shift, int C) {
  int c = blockIdx.x * blockDim.x + threadIdx.x;
  if (c < C) {
    float s = g[c] * rsqrtf(vr[c] + 1e-5f);
    scale[c] = s;
    float b0 = bias ? bias[c] : 0.f;
    shift[c] = (b0 - mn[c]) * s + bt[c];
  }
}

// torch ConvTranspose2d weight [CI, CO, 4, 4] -> conv weight [CO, CI*16] (flipped)
__global__ void repack_deconv(const float* __restrict__ w, float* __restrict__ out,
                              int CI, int CO) {
  int i = blockIdx.x * blockDim.x + threadIdx.x;
  int total = CI * CO * 16;
  if (i >= total) return;
  int kw = i & 3;
  int kh = (i >> 2) & 3;
  int t = i >> 4;
  int co = t % CO;
  int ci = t / CO;
  out[((long)co * CI + ci) * 16 + kh * 4 + kw] =
      w[(((long)ci * CO + co) * 4 + (3 - kh)) * 4 + (3 - kw)];
}

__global__ __launch_bounds__(256)
void softmax_rows(float* __restrict__ s, int rows, int L) {
  int wave = threadIdx.x >> 5;
  int lane = threadIdx.x & 31;
  int row = blockIdx.x * 8 + wave;
  if (row >= rows) return;
  float* p = s + (long)row * L;
  float mx = -1e30f;
  for (int i = lane; i < L; i += 32) mx = fmaxf(mx, p[i]);
  for (int o = 16; o; o >>= 1) mx = fmaxf(mx, __shfl_xor(mx, o, 32));
  float sum = 0.f;
  for (int i = lane; i < L; i += 32) { float e = __expf(p[i] - mx); p[i] = e; sum += e; }
  for (int o = 16; o; o >>= 1) sum += __shfl_xor(sum, o, 32);
  float inv = 1.0f / sum;
  for (int i = lane; i < L; i += 32) p[i] *= inv;
}

// adaptive avg pool NCHW [B,C,H,W] -> [B, C*16] flat (c*16 + h4*4 + w4)
__global__ void avgpool_4x4(const float* __restrict__ in, float* __restrict__ out,
                            int B, int C, int H, int W) {
  int idx = blockIdx.x * blockDim.x + threadIdx.x;
  int total = B * C * 16;
  if (idx >= total) return;
  int w4 = idx & 3;
  int h4 = (idx >> 2) & 3;
  int t = idx >> 4;
  int c = t % C;
  int b = t / C;
  int bh = H / 4, bw = W / 4;
  const float* p = in + ((long)b * C + c) * H * W;
  float sum = 0.f;
  for (int i = 0; i < bh; ++i)
    for (int j = 0; j < bw; ++j) sum += p[(h4 * bh + i) * W + (w4 * bw + j)];
  out[(long)b * C * 16 + c * 16 + h4 * 4 + w4] = sum / (float)(bh * bw);
}

__global__ void resize_bilinear(const float* __restrict__ in, float* __restrict__ out,
                                int B, int C, int IH, int IW, int OH, int OW) {
  long idx = (long)blockIdx.x * blockDim.x + threadIdx.x;
  long total = (long)B * C * OH * OW;
  if (idx >= total) return;
  int ow = (int)(idx % OW);
  long t = idx / OW;
  int oh = (int)(t % OH);
  t /= OH;
  int c = (int)(t % C);
  int b = (int)(t / C);
  float sh = (float)IH / (float)OH, sw = (float)IW / (float)OW;
  float fy = (oh + 0.5f) * sh - 0.5f;
  float fx = (ow + 0.5f) * sw - 0.5f;
  int y0 = (int)floorf(fy); float dy = fy - y0;
  int x0 = (int)floorf(fx); float dx = fx - x0;
  int y1 = imin_(imax_(y0 + 1, 0), IH - 1);
  int x1 = imin_(imax_(x0 + 1, 0), IW - 1);
  y0 = imin_(imax_(y0, 0), IH - 1);
  x0 = imin_(imax_(x0, 0), IW - 1);
  const float* p = in + ((long)b * C + c) * IH * IW;
  float v = (1.f - dy) * ((1.f - dx) * p[y0 * IW + x0] + dx * p[y0 * IW + x1]) +
            dy * ((1.f - dx) * p[y1 * IW + x0] + dx * p[y1 * IW + x1]);
  out[idx] = v;
}

// VQ: one block per batch row; z [B,D] row-major, codebook [NC,D]
__global__ __launch_bounds__(256)
void vq_assign(const float* __restrict__ z, const float* __restrict__ cb,
               float* __restrict__ zq, float* __restrict__ loss_acc,
               float* __restrict__ out_inds, int B, int NC, int D) {
  __shared__ float zl[256];
  __shared__ float bestd[256];
  __shared__ int besti[256];
  int b = blockIdx.x;
  int tid = threadIdx.x;
  for (int d = tid; d < D; d += 256) zl[d] = z[(long)b * D + d];
  __syncthreads();
  float bd = 1e30f;
  int bi = 0;
  for (int j = tid; j < NC; j += 256) {
    const float* c = cb + (long)j * D;
    float acc = 0.f;
    for (int d = 0; d < D; ++d) { float df = zl[d] - c[d]; acc += df * df; }
    if (acc < bd) { bd = acc; bi = j; }
  }
  bestd[tid] = bd;
  besti[tid] = bi;
  __syncthreads();
  for (int s = 128; s; s >>= 1) {
    if (tid < s) {
      if (bestd[tid + s] < bestd[tid] ||
          (bestd[tid + s] == bestd[tid] && besti[tid + s] < besti[tid])) {
        bestd[tid] = bestd[tid + s];
        besti[tid] = besti[tid + s];
      }
    }
    __syncthreads();
  }
  int best = besti[0];
  float part = 0.f;
  const float* c = cb + (long)best * D;
  for (int d = tid; d < D; d += 256) {
    float q = c[d];
    zq[(long)b * D + d] = q;
    float df = q - zl[d];
    part += df * df;
  }
  __syncthreads();
  bestd[tid] = part;
  __syncthreads();
  for (int s = 128; s; s >>= 1) {
    if (tid < s) bestd[tid] += bestd[tid + s];
    __syncthreads();
  }
  if (tid == 0) {
    atomicAdd(loss_acc, bestd[0]);
    out_inds[b] = (float)best;
  }
}

__global__ void vq_finalize(const float* __restrict__ acc, float* __restrict__ out) {
  out[0] = acc[0] * 1.25f / (32.0f * 256.0f);
}

// ---------------------------------------------------------------------------
// Parameter unpacking (both insertion-order and sorted-key-order flattening)
// ---------------------------------------------------------------------------
struct BNP { const float *g, *b, *m, *v; };
struct ResP { const float *w1, *b1, *w2, *b2; BNP bn1, bn2; };
struct AttnP { const float *wq, *bq, *wk, *bk, *wv, *bv, *gamma; };
struct LinP { const float *w, *b; };
struct Net {
  const float* x;
  const float* codebook;
  const float *c1w, *c1b, *c2w, *c2b, *c3w, *c3b, *c4w, *c4b;
  BNP ebn1, ebn2, ebn3, ebn4;
  ResP eres1, eres2, eres3;
  AttnP eattn;
  LinP el1, el2, el3;
  LinP dl1, dl2, dl3;
  AttnP dattn;
  const float *d1w, *d1b, *d2w, *d2b, *d3w, *d3b, *d4w, *d4b;
  BNP dbn1, dbn2, dbn3, dbn4;
  ResP dres1, dres2, dres3;
  const float *s1w, *s1b, *s2w, *s2b, *s3w, *s3b;
  const float *fw, *fb;
};

static void unpack_insertion(void* const* d_in, Net& n) {
  int i = 0;
  auto nx = [&]() { return (const float*)d_in[i++]; };
  auto bn = [&](BNP& b) { b.g = nx(); b.b = nx(); b.m = nx(); b.v = nx(); };
  auto res = [&](ResP& r) { r.w1 = nx(); r.b1 = nx(); bn(r.bn1); r.w2 = nx(); r.b2 = nx(); bn(r.bn2); };
  auto attn = [&](AttnP& a) { a.wq = nx(); a.bq = nx(); a.wk = nx(); a.bk = nx(); a.wv = nx(); a.bv = nx(); a.gamma = nx(); };
  auto lin = [&](LinP& l) { l.w = nx(); l.b = nx(); };
  n.x = nx();
  n.c1w = nx(); n.c1b = nx(); bn(n.ebn1); res(n.eres1);
  n.c2w = nx(); n.c2b = nx(); bn(n.ebn2); res(n.eres2);
  n.c3w = nx(); n.c3b = nx(); bn(n.ebn3); res(n.eres3);
  n.c4w = nx(); n.c4b = nx(); bn(n.ebn4);
  attn(n.eattn); lin(n.el1); lin(n.el2); lin(n.el3);
  lin(n.dl1); lin(n.dl2); lin(n.dl3); attn(n.dattn);
  n.d1w = nx(); n.d1b = nx(); bn(n.dbn1); n.s1w = nx(); n.s1b = nx(); res(n.dres1);
  n.d2w = nx(); n.d2b = nx(); bn(n.dbn2); n.s2w = nx(); n.s2b = nx(); res(n.dres2);
  n.d3w = nx(); n.d3b = nx(); bn(n.dbn3); n.s3w = nx(); n.s3b = nx(); res(n.dres3);
  n.d4w = nx(); n.d4b = nx(); bn(n.dbn4); n.fw = nx(); n.fb = nx();
  n.codebook = nx();
}

static void unpack_sorted(void* const* d_in, Net& n) {
  int i = 0;
  auto nx = [&]() { return (const float*)d_in[i++]; };
  auto bn = [&](BNP& b) { b.b = nx(); b.g = nx(); b.m = nx(); b.v = nx(); };  // beta,gamma,mean,var
  auto res = [&](ResP& r) { r.b1 = nx(); r.b2 = nx(); bn(r.bn1); bn(r.bn2); r.w1 = nx(); r.w2 = nx(); };
  auto attn = [&](AttnP& a) { a.bk = nx(); a.bq = nx(); a.bv = nx(); a.gamma = nx(); a.wk = nx(); a.wq = nx(); a.wv = nx(); };
  auto lin = [&](LinP& l) { l.b = nx(); l.w = nx(); };
  n.codebook = nx();
  attn(n.dattn); bn(n.dbn1); bn(n.dbn2); bn(n.dbn3); bn(n.dbn4);
  n.d1b = nx(); n.d1w = nx(); n.d2b = nx(); n.d2w = nx();
  n.d3b = nx(); n.d3w = nx(); n.d4b = nx(); n.d4w = nx();
  n.fb = nx(); n.fw = nx();
  lin(n.dl1); lin(n.dl2); lin(n.dl3);
  res(n.dres1); res(n.dres2); res(n.dres3);
  n.s1b = nx(); n.s1w = nx(); n.s2b = nx(); n.s2w = nx(); n.s3b = nx(); n.s3w = nx();
  attn(n.eattn); bn(n.ebn1); bn(n.ebn2); bn(n.ebn3); bn(n.ebn4);
  n.c1b = nx(); n.c1w = nx(); n.c2b = nx(); n.c2w = nx();
  n.c3b = nx(); n.c3w = nx(); n.c4b = nx(); n.c4w = nx();
  lin(n.el1); lin(n.el2); lin(n.el3);
  res(n.eres1); res(n.eres2); res(n.eres3);
  n.x = nx();
}

// ---------------------------------------------------------------------------
// Driver
// ---------------------------------------------------------------------------
extern "C" void kernel_launch(void* const* d_in, const int* in_sizes, int n_in,
                              void* d_out, int out_size, void* d_ws, size_t ws_size,
                              hipStream_t stream) {
  (void)n_in; (void)out_size; (void)ws_size;
  Net net;
  if (in_sizes[0] == 32 * 4 * 256 * 256) unpack_insertion(d_in, net);
  else unpack_sorted(d_in, net);   // jax-tree sorted-key flattening (codebook first)

  float* recon = (float*)d_out;
  const long RECON = 32L * 4 * 256 * 256;  // 8,388,608

  // ---- workspace arena (fp32 elems) ----
  float* ws = (float*)d_ws;
  size_t off = 0;
  auto A_ = [&](size_t nelem) { float* p = ws + off; off += (nelem + 63) & ~(size_t)63; return p; };
  float* SC = A_(256);      float* SH = A_(256);
  float* WP = A_(524288);   float* LOSS = A_(64);
  float* E1 = A_(16777216); float* T1 = A_(16777216); float* R1 = A_(16777216);
  float* E2 = A_(8388608);  float* T2 = A_(8388608);  float* R2 = A_(8388608);
  float* E3 = A_(4194304);  float* T3 = A_(4194304);  float* R3 = A_(4194304);
  float* E4 = A_(2097152);
  float* Qb = A_(262144);   float* Kb = A_(262144);   float* Vb = A_(2097152);
  float* Sb = A_(2097152);  float* A4 = A_(2097152);
  float* P  = A_(131072);
  float* F1 = A_(32768);    float* F2 = A_(16384);
  float* Z  = A_(8192);     float* ZQ = A_(8192);
  float* G1 = A_(16384);    float* G2 = A_(32768);    float* G3 = A_(131072);
  float* DQ = A_(16384);    float* DK = A_(16384);    float* DV = A_(131072);
  float* DS2 = A_(8192);    float* DA = A_(131072);
  float* X1a = A_(262144);  float* X1b = A_(262144);  float* X1c = A_(262144);  float* SK1 = A_(262144);
  float* X2a = A_(524288);  float* X2b = A_(524288);  float* X2c = A_(524288);  float* SK2 = A_(524288);
  float* X3a = A_(1048576); float* X3b = A_(1048576); float* X3c = A_(1048576); float* SK3 = A_(1048576);
  float* X4 = A_(2097152);
  float* R256 = E1;  // reuse E1+T1 (exactly 33,554,432 elems), dead by decoder tail

  // ---- host-side launch helpers ----
  auto gemm = [&](const float* A, long abs_, long asm_, long ask,
                  const float* B, long bbs, long bsk, long bsn,
                  float* O, long obs, long osm, long osn,
                  const float* scale, const float* shift, const float* resid,
                  const float* alpha, int M, int N, int K, int relu, int batches) {
    dim3 g((N + BN - 1) / BN, (M + BM - 1) / BM, batches), bl(NT);
    gemm_bf16_wmma<<<g, bl, 0, stream>>>(A, abs_, asm_, ask, B, bbs, bsk, bsn,
                                         O, obs, osm, osn, scale, shift, resid, alpha,
                                         M, N, K, relu);
  };
  auto conv = [&](const float* X, const float* Wt, const float* scale, const float* shift,
                  const float* resid, float* Out, int IC, int IH, int IW,
                  int OC, int OH, int OW, int KH, int KW, int strd, int pad, int ldil, int relu) {
    int N = 32 * OH * OW;
    dim3 g((N + BN - 1) / BN, (OC + BM - 1) / BM, 1), bl(NT);
    conv_bf16_wmma<<<g, bl, 0, stream>>>(X, Wt, scale, shift, resid, Out,
                                         32, IC, IH, IW, OC, OH, OW,
                                         KH, KW, strd, pad, ldil, relu);
  };
  auto fold = [&](const BNP& bn, const float* bias, int C) {
    fold_bn_kernel<<<1, 256, 0, stream>>>(bn.g, bn.b, bn.m, bn.v, bias, SC, SH, C);
  };
  auto resblock = [&](const float* xin, float* t, float* yout, const ResP& r,
                      int C, int H, int W) {
    fold(r.bn1, r.b1, C);
    conv(xin, r.w1, SC, SH, nullptr, t, C, H, W, C, H, W, 3, 3, 1, 1, 1, 1);
    fold(r.bn2, r.b2, C);
    conv(t, r.w2, SC, SH, xin, yout, C, H, W, C, H, W, 3, 3, 1, 1, 1, 1);
  };
  auto attention = [&](const float* x, int C, int H, int W, const AttnP& a,
                       float* q, float* k, float* v, float* s, float* outp) {
    int HW = H * W, Cq = C / 8;
    conv(x, a.wq, nullptr, a.bq, nullptr, q, C, H, W, Cq, H, W, 1, 1, 1, 0, 1, 0);
    conv(x, a.wk, nullptr, a.bk, nullptr, k, C, H, W, Cq, H, W, 1, 1, 1, 0, 1, 0);
    conv(x, a.wv, nullptr, a.bv, nullptr, v, C, H, W, C, H, W, 1, 1, 1, 0, 1, 0);
    // s[b,n,m] = sum_c q[b,c,n] * k[b,c,m]
    gemm(q, (long)Cq * HW, 1, HW, k, (long)Cq * HW, HW, 1,
         s, (long)HW * HW, HW, 1, nullptr, nullptr, nullptr, nullptr,
         HW, HW, Cq, 0, 32);
    int rows = 32 * HW;
    softmax_rows<<<(rows + 7) / 8, 256, 0, stream>>>(s, rows, HW);
    // out[b,c,n] = gamma * sum_m v[b,c,m] * s[b,n,m] + x
    gemm(v, (long)C * HW, HW, 1, s, (long)HW * HW, 1, HW,
         outp, (long)C * HW, HW, 1, nullptr, nullptr, x, a.gamma,
         C, HW, HW, 0, 32);
  };
  auto linear = [&](const LinP& l, const float* xin, float* outp, int M, int Kd, int relu) {
    gemm(l.w, 0, Kd, 1, xin, 0, 1, Kd, outp, 0, 1, M,
         nullptr, l.b, nullptr, nullptr, M, 32, Kd, relu, 1);
  };
  auto deconv = [&](const float* xin, const float* wdec, const float* bias, const BNP& bn,
                    float* outp, int CI, int CO, int IH, int IW) {
    int tot = CI * CO * 16;
    repack_deconv<<<(tot + 255) / 256, 256, 0, stream>>>(wdec, WP, CI, CO);
    fold(bn, bias, CO);
    conv(xin, WP, SC, SH, nullptr, outp, CI, IH, IW, CO, 2 * IH, 2 * IW, 4, 4, 1, 2, 2, 1);
  };
  auto resize = [&](const float* in, float* outp, int C, int IH, int IW, int OH, int OW) {
    long tot = 32L * C * OH * OW;
    resize_bilinear<<<(int)((tot + 255) / 256), 256, 0, stream>>>(in, outp, 32, C, IH, IW, OH, OW);
  };

  // ================= ENCODER =================
  fold(net.ebn1, net.c1b, 32);
  conv(net.x, net.c1w, SC, SH, nullptr, E1, 4, 256, 256, 32, 128, 128, 7, 7, 2, 3, 1, 1);
  resblock(E1, T1, R1, net.eres1, 32, 128, 128);            // skip0 = R1
  fold(net.ebn2, net.c2b, 64);
  conv(R1, net.c2w, SC, SH, nullptr, E2, 32, 128, 128, 64, 64, 64, 3, 3, 2, 1, 1, 1);
  resblock(E2, T2, R2, net.eres2, 64, 64, 64);              // skip1 = R2
  fold(net.ebn3, net.c3b, 128);
  conv(R2, net.c3w, SC, SH, nullptr, E3, 64, 64, 64, 128, 32, 32, 3, 3, 2, 1, 1, 1);
  resblock(E3, T3, R3, net.eres3, 128, 32, 32);             // skip2 = R3
  fold(net.ebn4, net.c4b, 256);
  conv(R3, net.c4w, SC, SH, nullptr, E4, 128, 32, 32, 256, 16, 16, 3, 3, 2, 1, 1, 1);
  attention(E4, 256, 16, 16, net.eattn, Qb, Kb, Vb, Sb, A4);
  avgpool_4x4<<<(32 * 256 * 16 + 255) / 256, 256, 0, stream>>>(A4, P, 32, 256, 16, 16);
  linear(net.el1, P, F1, 1024, 4096, 1);
  linear(net.el2, F1, F2, 512, 1024, 1);
  linear(net.el3, F2, Z, 256, 512, 0);

  // ================= VQ =================
  zero_kernel<<<1, 64, 0, stream>>>(LOSS, 64);
  vq_assign<<<32, 256, 0, stream>>>(Z, net.codebook, ZQ, LOSS, recon + RECON + 1, 32, 512, 256);
  vq_finalize<<<1, 1, 0, stream>>>(LOSS, recon + RECON);

  // ================= DECODER =================
  linear(net.dl1, ZQ, G1, 512, 256, 1);
  linear(net.dl2, G1, G2, 1024, 512, 1);
  linear(net.dl3, G2, G3, 4096, 1024, 1);                   // [32,256,4,4] NCHW flat
  attention(G3, 256, 4, 4, net.dattn, DQ, DK, DV, DS2, DA);
  deconv(DA, net.d1w, net.d1b, net.dbn1, X1a, 256, 128, 4, 4);   // -> 8x8
  resize(R3, SK1, 128, 32, 32, 8, 8);
  conv(SK1, net.s1w, nullptr, net.s1b, X1a, X1b, 128, 8, 8, 128, 8, 8, 1, 1, 1, 0, 1, 0);
  resblock(X1b, X1c, X1a, net.dres1, 128, 8, 8);
  deconv(X1a, net.d2w, net.d2b, net.dbn2, X2a, 128, 64, 8, 8);   // -> 16x16
  resize(R2, SK2, 64, 64, 64, 16, 16);
  conv(SK2, net.s2w, nullptr, net.s2b, X2a, X2b, 64, 16, 16, 64, 16, 16, 1, 1, 1, 0, 1, 0);
  resblock(X2b, X2c, X2a, net.dres2, 64, 16, 16);
  deconv(X2a, net.d3w, net.d3b, net.dbn3, X3a, 64, 32, 16, 16);  // -> 32x32
  resize(R1, SK3, 32, 128, 128, 32, 32);
  conv(SK3, net.s3w, nullptr, net.s3b, X3a, X3b, 32, 32, 32, 32, 32, 32, 1, 1, 1, 0, 1, 0);
  resblock(X3b, X3c, X3a, net.dres3, 32, 32, 32);
  deconv(X3a, net.d4w, net.d4b, net.dbn4, X4, 32, 16, 32, 32);   // -> 64x64
  resize(X4, R256, 16, 64, 64, 256, 256);
  conv(R256, net.fw, nullptr, net.fb, nullptr, recon, 16, 256, 256, 4, 256, 256, 3, 3, 1, 1, 1, 0);
}
